// GNN_GRU_Model_16123307229305
// MI455X (gfx1250) — compile-verified, matrix operand order checked by reading, not run on hardware
//
#include <hip/hip_runtime.h>
#include <hip/hip_bf16.h>

typedef __attribute__((ext_vector_type(16))) _Float16 v16h;
typedef __attribute__((ext_vector_type(8)))  _Float16 v8h;
typedef __attribute__((ext_vector_type(8)))  float    v8f;

#define T_STEPS 5
#define NTOT    16384
#define E_EDGES 131072
#define CIN     64
#define HDIM    128
#define NAG     1024
#define GDIM    384   // 3*H for GRU gates

union fragH { v16h f; v8h h[2]; };

// ---------------- elementwise / setup kernels ----------------

__global__ void fill_kernel(float* p, float v, int n) {
    int i = blockIdx.x * blockDim.x + threadIdx.x;
    if (i < n) p[i] = v;
}

__global__ void f32_to_f16_kernel(const float* __restrict__ in, _Float16* __restrict__ out, int n) {
    int i = blockIdx.x * blockDim.x + threadIdx.x;
    if (i < n) out[i] = (_Float16)in[i];
}

// Pack an f32 weight matrix into the per-lane WMMA B-fragment layout (f16).
// Logical B is K x N (row-major); trans!=0 means B[k][n] = W[n*K + k] (for x @ W^T).
// Packed: Bp[ ((nt*KC + kc)*32 + hh*16 + nn)*16 + j ], KC = K/32,
//   kk = k%32: hh = (kk>>3)&1, j = (kk&7) + ((kk>>4)<<3)   (matches A/B lane layout)
__global__ void pack_b_f32_kernel(const float* __restrict__ W, _Float16* __restrict__ Bp,
                                  int K, int N, int trans) {
    int i = blockIdx.x * blockDim.x + threadIdx.x;
    if (i >= K * N) return;
    int k = i / N, n = i % N;
    float v = trans ? W[(size_t)n * K + k] : W[(size_t)k * N + n];
    int kc = k >> 5, kk = k & 31;
    int nt = n >> 4, nn = n & 15;
    int hh = (kk >> 3) & 1;
    int j  = (kk & 7) + ((kk >> 4) << 3);
    Bp[((size_t)(nt * (K >> 5) + kc) * 32 + hh * 16 + nn) * 16 + j] = (_Float16)v;
}

__global__ void deg_accum_kernel(const int* __restrict__ tgt, float* __restrict__ deg, int n) {
    int i = blockIdx.x * blockDim.x + threadIdx.x;
    if (i < n) atomicAdd(&deg[tgt[i]], 1.0f);
}

// deg >= 1 always (self loop), so dis = rsqrt(deg) exactly
__global__ void deg_to_dis_kernel(float* deg, int n) {
    int i = blockIdx.x * blockDim.x + threadIdx.x;
    if (i < n) deg[i] = rsqrtf(deg[i]);
}

// ---------------- WMMA GEMM: C(f32, MxN) = A(f16, MxK row-major) @ Bp(packed) ----------------
// 256 threads = 8 waves; each wave owns a 16x(16*NT) strip of C. K, N compile-time.
template <int N, int K, int NT>
__global__ void gemm_wmma_kernel(const _Float16* __restrict__ A,
                                 const _Float16* __restrict__ Bp,
                                 float* __restrict__ C, int M) {
    const int wave = threadIdx.x >> 5;
    const int lane = threadIdx.x & 31;
    const int m0  = (blockIdx.x * 8 + wave) * 16;
    const int nt0 = blockIdx.y * NT;
    if (m0 >= M) return;                  // wave-uniform: EXEC all-ones at WMMA
    const int hh  = lane >> 4;
    const int l16 = lane & 15;
    const int kb  = hh * 8;
    constexpr int KC = K >> 5;
    v8f acc[NT] = {};
#pragma unroll
    for (int kc = 0; kc < KC; ++kc) {
        // A fragment: row m0+l16; halves [kb..kb+7] and [kb+16..kb+23] — two b128 loads
        fragH a;
        const _Float16* ap = A + (size_t)(m0 + l16) * K + kc * 32 + kb;
        a.h[0] = *(const v8h*)(ap);
        a.h[1] = *(const v8h*)(ap + 16);
#pragma unroll
        for (int t = 0; t < NT; ++t) {
            // B fragment: pre-packed, contiguous 32B per lane — two b128 loads
            fragH b;
            const _Float16* bp = Bp + ((size_t)((nt0 + t) * KC + kc) * 32 + lane) * 16;
            b.h[0] = *(const v8h*)(bp);
            b.h[1] = *(const v8h*)(bp + 8);
            acc[t] = __builtin_amdgcn_wmma_f32_16x16x32_f16(false, a.f, false, b.f,
                                                            (short)0, acc[t], false, false);
        }
    }
#pragma unroll
    for (int t = 0; t < NT; ++t) {
        // C/D layout: VGPR r -> row m0 + r + hh*8, col n0 + l16
        float* cp = C + (size_t)(m0 + hh * 8) * N + (nt0 + t) * 16 + l16;
#pragma unroll
        for (int r = 0; r < 8; ++r) cp[(size_t)r * N] = acc[t][r];
    }
}

// ---------------- edge scatter: agg[tgt] += hw[src] * norm (+ self loops) ----------------
// One wave per edge-unit; lane handles 4 contiguous channels of 128.
__global__ void scatter_kernel(const float* __restrict__ hw,
                               const int* __restrict__ src,
                               const int* __restrict__ tgt,
                               const float* __restrict__ dis,
                               float* __restrict__ agg) {
    const int unit = blockIdx.x * (blockDim.x >> 5) + (threadIdx.x >> 5);
    const int lane = threadIdx.x & 31;
    if (unit >= E_EDGES + NTOT) return;
    int s, t; float norm;
    if (unit < E_EDGES) {
        s = src[unit]; t = tgt[unit];
        norm = dis[s] * dis[t];
    } else {
        s = t = unit - E_EDGES;          // self loop
        float d = dis[s]; norm = d * d;
    }
    const float4 v = *(const float4*)(hw + (size_t)s * HDIM + lane * 4);
    float* out = agg + (size_t)t * HDIM + lane * 4;
    atomicAdd(out + 0, v.x * norm);
    atomicAdd(out + 1, v.y * norm);
    atomicAdd(out + 2, v.z * norm);
    atomicAdd(out + 3, v.w * norm);
}

__global__ void bias_relu_f16_kernel(const float* __restrict__ agg, const float* __restrict__ b,
                                     _Float16* __restrict__ out, int n) {
    int i = blockIdx.x * blockDim.x + threadIdx.x;
    if (i < n) {
        float v = agg[i] + b[i & (HDIM - 1)];
        out[i] = (_Float16)(v > 0.f ? v : 0.f);
    }
}

// layer-2 output only needed at agent rows: fuse bias+relu+gather+f16 pack
__global__ void agent_gather_kernel(const float* __restrict__ agg, const float* __restrict__ b2,
                                    _Float16* __restrict__ ag) {
    int i = blockIdx.x * blockDim.x + threadIdx.x;
    if (i < NAG * HDIM) {
        int j = i >> 7, c = i & (HDIM - 1);
        int node = (j >> 4) * 256 + (j & 15);      // agent_idx = b*N_PER + a
        float v = agg[(size_t)node * HDIM + c] + b2[c];
        ag[i] = (_Float16)(v > 0.f ? v : 0.f);
    }
}

// ---------------- GRU gates ----------------
__global__ void gru_gate_kernel(const float* __restrict__ gi, const float* __restrict__ gh,
                                const float* __restrict__ b_ih, const float* __restrict__ b_hh,
                                float* __restrict__ h) {
    int i = blockIdx.x * blockDim.x + threadIdx.x;
    if (i >= NAG * HDIM) return;
    int j = i >> 7, c = i & (HDIM - 1);
    const float* gij = gi + (size_t)j * GDIM;
    const float* ghj = gh + (size_t)j * GDIM;
    float ir  = gij[c]            + b_ih[c];
    float iz  = gij[HDIM + c]     + b_ih[HDIM + c];
    float inn = gij[2 * HDIM + c] + b_ih[2 * HDIM + c];
    float hr  = ghj[c]            + b_hh[c];
    float hz  = ghj[HDIM + c]     + b_hh[HDIM + c];
    float hn  = ghj[2 * HDIM + c] + b_hh[2 * HDIM + c];
    float r = 1.f / (1.f + expf(-(ir + hr)));
    float z = 1.f / (1.f + expf(-(iz + hz)));
    float nn = tanhf(inn + r * hn);
    h[i] = (1.f - z) * nn + z * h[i];
}

__global__ void final_linear_kernel(const float* __restrict__ h, const float* __restrict__ w,
                                    const float* __restrict__ b, float* __restrict__ out) {
    int i = blockIdx.x * blockDim.x + threadIdx.x;
    if (i >= NAG * 2) return;
    int j = i >> 1, c = i & 1;
    float s = b[c];
    for (int k = 0; k < HDIM; ++k) s += h[(size_t)j * HDIM + k] * w[k * 2 + c];
    out[i] = s;   // (B, A, COUT) flat == row j = b*16+a
}

// ---------------- host orchestration ----------------

extern "C" void kernel_launch(void* const* d_in, const int* in_sizes, int n_in,
                              void* d_out, int out_size, void* d_ws, size_t ws_size,
                              hipStream_t stream) {
    (void)in_sizes; (void)n_in; (void)out_size; (void)ws_size;
    const float* x     = (const float*)d_in[0];
    const int*   ei    = (const int*)  d_in[1];
    const float* W1    = (const float*)d_in[2];
    const float* b1    = (const float*)d_in[3];
    const float* W2    = (const float*)d_in[4];
    const float* b2    = (const float*)d_in[5];
    const float* w_ih  = (const float*)d_in[6];
    const float* w_hh  = (const float*)d_in[7];
    const float* b_ih  = (const float*)d_in[8];
    const float* b_hh  = (const float*)d_in[9];
    const float* w_lin = (const float*)d_in[10];
    const float* b_lin = (const float*)d_in[11];

    char* ws = (char*)d_ws;
    size_t off = 0;
    auto alloc = [&](size_t bytes) -> void* {
        void* p = ws + off;
        off = (off + bytes + 255) & ~((size_t)255);
        return p;
    };
    float*    deg   = (float*)   alloc((size_t)NTOT * 4);
    _Float16* h16   = (_Float16*)alloc((size_t)NTOT * HDIM * 2);
    float*    hw    = (float*)   alloc((size_t)NTOT * HDIM * 4);
    float*    agg   = (float*)   alloc((size_t)NTOT * HDIM * 4);
    _Float16* W1p   = (_Float16*)alloc((size_t)CIN * HDIM * 2);
    _Float16* W2p   = (_Float16*)alloc((size_t)HDIM * HDIM * 2);
    _Float16* wihp  = (_Float16*)alloc((size_t)HDIM * GDIM * 2);
    _Float16* whhp  = (_Float16*)alloc((size_t)HDIM * GDIM * 2);
    _Float16* ag16  = (_Float16*)alloc((size_t)T_STEPS * NAG * HDIM * 2);
    float*    hst   = (float*)   alloc((size_t)NAG * HDIM * 4);
    _Float16* hst16 = (_Float16*)alloc((size_t)NAG * HDIM * 2);
    float*    gi    = (float*)   alloc((size_t)NAG * GDIM * 4);
    float*    gh    = (float*)   alloc((size_t)NAG * GDIM * 4);

    const int TB = 256;
    // weights -> packed f16 WMMA B fragments (transposed for x @ W^T in the GRU)
    pack_b_f32_kernel<<<(CIN * HDIM + TB - 1) / TB, TB, 0, stream>>>(W1, W1p, CIN, HDIM, 0);
    pack_b_f32_kernel<<<(HDIM * HDIM + TB - 1) / TB, TB, 0, stream>>>(W2, W2p, HDIM, HDIM, 0);
    pack_b_f32_kernel<<<(HDIM * GDIM + TB - 1) / TB, TB, 0, stream>>>(w_ih, wihp, HDIM, GDIM, 1);
    pack_b_f32_kernel<<<(HDIM * GDIM + TB - 1) / TB, TB, 0, stream>>>(w_hh, whhp, HDIM, GDIM, 1);

    dim3 gGcn(NTOT / 16 / 8, HDIM / 16 / 2);      // (128, 4)
    const int scatterBlocks = (E_EDGES + NTOT) / 8;

    for (int t = 0; t < T_STEPS; ++t) {
        const int* src = ei + (size_t)t * 2 * E_EDGES;
        const int* tgt = src + E_EDGES;
        // degree -> dis (self loop baked in as deg=1 init)
        fill_kernel<<<NTOT / TB, TB, 0, stream>>>(deg, 1.0f, NTOT);
        deg_accum_kernel<<<E_EDGES / TB, TB, 0, stream>>>(tgt, deg, E_EDGES);
        deg_to_dis_kernel<<<NTOT / TB, TB, 0, stream>>>(deg, NTOT);
        // ---- GCN layer 1 ----
        f32_to_f16_kernel<<<(NTOT * CIN) / TB, TB, 0, stream>>>(x + (size_t)t * NTOT * CIN, h16, NTOT * CIN);
        gemm_wmma_kernel<HDIM, CIN, 2><<<gGcn, 256, 0, stream>>>(h16, W1p, hw, NTOT);
        fill_kernel<<<(NTOT * HDIM) / TB, TB, 0, stream>>>(agg, 0.0f, NTOT * HDIM);
        scatter_kernel<<<scatterBlocks, 256, 0, stream>>>(hw, src, tgt, deg, agg);
        bias_relu_f16_kernel<<<(NTOT * HDIM) / TB, TB, 0, stream>>>(agg, b1, h16, NTOT * HDIM);
        // ---- GCN layer 2 ----
        gemm_wmma_kernel<HDIM, HDIM, 2><<<gGcn, 256, 0, stream>>>(h16, W2p, hw, NTOT);
        fill_kernel<<<(NTOT * HDIM) / TB, TB, 0, stream>>>(agg, 0.0f, NTOT * HDIM);
        scatter_kernel<<<scatterBlocks, 256, 0, stream>>>(hw, src, tgt, deg, agg);
        agent_gather_kernel<<<(NAG * HDIM) / TB, TB, 0, stream>>>(agg, b2, ag16 + (size_t)t * NAG * HDIM);
    }

    // ---- GRU over agents ----
    fill_kernel<<<(NAG * HDIM) / TB, TB, 0, stream>>>(hst, 0.0f, NAG * HDIM);
    dim3 gGru(NAG / 16 / 8, GDIM / 16 / 2);       // (8, 12)
    for (int s = 0; s < T_STEPS; ++s) {
        f32_to_f16_kernel<<<(NAG * HDIM) / TB, TB, 0, stream>>>(hst, hst16, NAG * HDIM);
        gemm_wmma_kernel<GDIM, HDIM, 2><<<gGru, 256, 0, stream>>>(ag16 + (size_t)s * NAG * HDIM, wihp, gi, NAG);
        gemm_wmma_kernel<GDIM, HDIM, 2><<<gGru, 256, 0, stream>>>(hst16, whhp, gh, NAG);
        gru_gate_kernel<<<(NAG * HDIM) / TB, TB, 0, stream>>>(gi, gh, b_ih, b_hh, hst);
    }
    final_linear_kernel<<<(NAG * 2 + TB - 1) / TB, TB, 0, stream>>>(hst, w_lin, b_lin, (float*)d_out);
}
